// CausalSelfAttention_87617332838455
// MI455X (gfx1250) — matrix operand order repaired and checked by reading, not compiled, not run on hardware
//
#include <hip/hip_runtime.h>
#include <hip/hip_bf16.h>

typedef __attribute__((ext_vector_type(16))) __bf16 v16bf;
typedef __attribute__((ext_vector_type(8)))  float  v8f;
typedef __attribute__((ext_vector_type(4)))  int    v4i;

typedef __attribute__((address_space(1))) v4i g_v4i;   // global AS pointer elem
typedef __attribute__((address_space(3))) v4i l_v4i;   // LDS AS pointer elem

#define N_EMBD   1024
#define N_HEAD   16
#define HEAD_DIM 64
#define SEQ      2048
#define BATCH    4
#define NROW     (BATCH * SEQ)   /* 8192 */

#define WMMA_BF16(a, b, c) \
    __builtin_amdgcn_wmma_f32_16x16x32_bf16(false, (a), false, (b), (short)0, (c), false, false)

union BFrag { v16bf bf; v4i i[2]; };

// Build a 16x32 (A) or 32x16 (B) bf16 fragment from two 16-byte chunks.
__device__ __forceinline__ v16bf load_frag2(const __bf16* p0, const __bf16* p1) {
    BFrag f;
    f.i[0] = *(const v4i*)p0;
    f.i[1] = *(const v4i*)p1;
    return f.bf;
}

// ---------------------------------------------------------------------------
// Async global -> LDS copy (16B per lane), CDNA5 ASYNCcnt-tracked path.
// ---------------------------------------------------------------------------
#if __has_builtin(__builtin_amdgcn_global_load_async_to_lds_b128)
__device__ __forceinline__ void async_cp16(const __bf16* g, __bf16* l) {
    __builtin_amdgcn_global_load_async_to_lds_b128(
        (g_v4i*)(void*)const_cast<__bf16*>(g),
        (l_v4i*)(void*)l, 0, 0);
}
#else
__device__ __forceinline__ void async_cp16(const __bf16* g, __bf16* l) {
    unsigned lo = (unsigned)(size_t)l;   // LDS byte offset (low 32 bits of generic ptr)
    asm volatile("global_load_async_to_lds_b128 %0, %1, off" :: "v"(lo), "v"(g) : "memory");
}
#endif

#if __has_builtin(__builtin_amdgcn_s_wait_asynccnt)
__device__ __forceinline__ void async_wait0() { __builtin_amdgcn_s_wait_asynccnt(0); }
#else
__device__ __forceinline__ void async_wait0() { asm volatile("s_wait_asynccnt 0" ::: "memory"); }
#endif

// ---------------------------------------------------------------------------
// fp32 -> bf16 conversion (grid-stride)
// ---------------------------------------------------------------------------
__global__ void csa_f32_to_bf16(const float* __restrict__ in, __bf16* __restrict__ out, int n) {
    int i = blockIdx.x * blockDim.x + threadIdx.x;
    int stride = gridDim.x * blockDim.x;
    for (; i < n; i += stride) out[i] = (__bf16)in[i];
}

// ---------------------------------------------------------------------------
// C[M,1024] = A[M,1024] @ W[1024,1024]^T   (both row-major, K contiguous)
// 4 waves/block; each wave computes a 16x64 tile of C (same n0 across waves).
// The shared 64x32 W k-tile is staged global->LDS with async copies, double
// buffered, so the copy of tile k+1 overlaps the WMMAs on tile k.
// mode 0: bf16 out [B,H,T,D] (scaled); mode 1: bf16 out [B,H,D,T]; mode 2: f32.
// ---------------------------------------------------------------------------
__global__ void __launch_bounds__(128)
csa_gemm_xwt(const __bf16* __restrict__ A, const __bf16* __restrict__ W,
             void* __restrict__ out, int mode, float scale) {
    __shared__ __bf16 Wl[2][64][32];   // 2 x 4KB double buffer

    const int lane = threadIdx.x & 31;
    const int half = lane >> 4;
    const int l16  = lane & 15;
    const int wave = threadIdx.x >> 5;
    const int m0 = (blockIdx.x * 4 + wave) * 16;
    const int n0 = blockIdx.y * 64;

    // Stage one 64x32 W tile: 256 16B chunks, 2 per thread.
    const int r0 = threadIdx.x >> 2,          c0 = (threadIdx.x & 3) * 8;
    const int r1 = (threadIdx.x + 128) >> 2,  c1 = ((threadIdx.x + 128) & 3) * 8;
#define STAGE_W(buf, kk)                                                   \
    do {                                                                   \
        async_cp16(W + (n0 + r0) * N_EMBD + (kk) + c0, &Wl[buf][r0][c0]);  \
        async_cp16(W + (n0 + r1) * N_EMBD + (kk) + c1, &Wl[buf][r1][c1]);  \
    } while (0)

    v8f acc[4] = {};

    // A fragment: lane l16 holds row m0+l16; lo chunk at k0 + 8*half, hi at +16.
    const __bf16* arow = A + (m0 + l16) * N_EMBD + (half ? 8 : 0);

    STAGE_W(0, 0);
    async_wait0();
    __syncthreads();

    for (int k0 = 0; k0 < N_EMBD; k0 += 32) {
        const int buf = (k0 >> 5) & 1;
        if (k0 + 32 < N_EMBD) STAGE_W(buf ^ 1, k0 + 32);   // overlap with WMMAs below

        __builtin_prefetch(arow + k0 + 128, 0, 0);
        v16bf a = load_frag2(arow + k0, arow + k0 + 16);
#pragma unroll
        for (int j = 0; j < 4; ++j) {
            // B fragment from LDS: lane holds row n0+j*16+l16; lo chunk at 8*half, hi at +16.
            const __bf16* wrow = &Wl[buf][j * 16 + l16][half ? 8 : 0];
            v16bf b = load_frag2(wrow, wrow + 16);
            acc[j] = WMMA_BF16(a, b, acc[j]);
        }

        async_wait0();      // next tile landed
        __syncthreads();    // everyone done reading buf before it is overwritten
    }
#undef STAGE_W

    if (mode == 2) {
        float* O = (float*)out;
#pragma unroll
        for (int j = 0; j < 4; ++j)
#pragma unroll
            for (int r = 0; r < 8; ++r) {
                int row = m0 + r + 8 * half;      // C-layout: vgpr r -> M = r (+8 for hi half)
                int col = n0 + j * 16 + l16;      // N = lane within half
                O[row * N_EMBD + col] = acc[j][r];
            }
    } else {
        __bf16* O = (__bf16*)out;
#pragma unroll
        for (int j = 0; j < 4; ++j)
#pragma unroll
            for (int r = 0; r < 8; ++r) {
                int row = m0 + r + 8 * half;      // = b*SEQ + t
                int col = n0 + j * 16 + l16;      // = h*HEAD_DIM + d
                int b = row >> 11, t = row & (SEQ - 1);
                int h = col >> 6,  d = col & (HEAD_DIM - 1);
                int idx;
                if (mode == 0) idx = ((b * N_HEAD + h) * SEQ + t) * HEAD_DIM + d;
                else           idx = ((b * N_HEAD + h) * HEAD_DIM + d) * SEQ + t;
                O[idx] = (__bf16)(acc[j][r] * scale);
            }
    }
}

// ---------------------------------------------------------------------------
// Flash attention: grid (B*H, T/64), 4 waves/block, each wave owns 16 q rows.
// Q,K: [B,H,T,D] bf16 (Q pre-scaled by 1/sqrt(D)); VT: [B,H,D,T] bf16.
// Ob: [B,T,H*D] bf16 attention output (pre output-projection).
// ---------------------------------------------------------------------------
__global__ void __launch_bounds__(128)
csa_flash_attn(const __bf16* __restrict__ Q, const __bf16* __restrict__ K,
               const __bf16* __restrict__ VT, __bf16* __restrict__ Ob) {
    __shared__ __bf16 Pl[4][16][32];   // per-wave 16x32 P tile (C-layout -> A-layout relay)

    const int lane = threadIdx.x & 31;
    const int half = lane >> 4;
    const int l16  = lane & 15;
    const int wave = threadIdx.x >> 5;
    const int bh = blockIdx.x;                       // 0..63
    const int t0 = blockIdx.y * 64 + wave * 16;
    const int b  = bh >> 4, h = bh & 15;

    const __bf16* q  = Q  + bh * (SEQ * HEAD_DIM);
    const __bf16* k  = K  + bh * (SEQ * HEAD_DIM);
    const __bf16* vt = VT + bh * (HEAD_DIM * SEQ);

    // Q fragments for d=[0,32) and d=[32,64)
    const __bf16* qrow = q + (t0 + l16) * HEAD_DIM + (half ? 16 : 0);
    v16bf aq0 = load_frag2(qrow,      qrow + 8);
    v16bf aq1 = load_frag2(qrow + 32, qrow + 40);

    v8f O0 = {}, O1 = {}, O2 = {}, O3 = {};
    float mrow[8], lrow[8];
#pragma unroll
    for (int r = 0; r < 8; ++r) { mrow[r] = -1e30f; lrow[r] = 0.f; }

    for (int s0 = 0; s0 <= t0 + 15; s0 += 32) {
        // ---- scores: S[16 x 32] as two 16x16 f32 tiles, K=64 split in two ----
        const __bf16* kr0 = k + (s0 + l16) * HEAD_DIM + (half ? 16 : 0);
        const __bf16* kr1 = kr0 + 16 * HEAD_DIM;
        v8f S0 = {}, S1 = {};
        S0 = WMMA_BF16(aq0, load_frag2(kr0,      kr0 + 8),  S0);
        S0 = WMMA_BF16(aq1, load_frag2(kr0 + 32, kr0 + 40), S0);
        S1 = WMMA_BF16(aq0, load_frag2(kr1,      kr1 + 8),  S1);
        S1 = WMMA_BF16(aq1, load_frag2(kr1 + 32, kr1 + 40), S1);

        // ---- causal mask ----
#pragma unroll
        for (int r = 0; r < 8; ++r) {
            int t = t0 + r + 8 * half;
            if (s0 + l16 > t)      S0[r] = -1e30f;
            if (s0 + 16 + l16 > t) S1[r] = -1e30f;
        }

        // ---- online softmax: rows live at (vgpr r, lane-half); reduce over 16 lanes ----
        float P0[8], P1[8];
#pragma unroll
        for (int r = 0; r < 8; ++r) {
            float mx = fmaxf(S0[r], S1[r]);
#pragma unroll
            for (int o = 8; o; o >>= 1) mx = fmaxf(mx, __shfl_xor(mx, o, 16));
            mx = fmaxf(mx, mrow[r]);
            float alpha = __expf(mrow[r] - mx);
            mrow[r] = mx;
            float p0 = __expf(S0[r] - mx);
            float p1 = __expf(S1[r] - mx);
            P0[r] = p0; P1[r] = p1;
            float sum = p0 + p1;
#pragma unroll
            for (int o = 8; o; o >>= 1) sum += __shfl_xor(sum, o, 16);
            lrow[r] = lrow[r] * alpha + sum;
            O0[r] *= alpha; O1[r] *= alpha; O2[r] *= alpha; O3[r] *= alpha;
        }

        // ---- relay P through LDS: C-layout -> A-layout (same wave, DS in-order) ----
#pragma unroll
        for (int r = 0; r < 8; ++r) {
            Pl[wave][r + 8 * half][l16]      = (__bf16)P0[r];
            Pl[wave][r + 8 * half][16 + l16] = (__bf16)P1[r];
        }
        const __bf16* pp = &Pl[wave][l16][half ? 16 : 0];
        v16bf ap = load_frag2(pp, pp + 8);

        // ---- O += P @ V : 4 d-tiles, B-fragments contiguous from transposed V ----
#pragma unroll
        for (int dt = 0; dt < 4; ++dt) {
            const __bf16* vr = vt + (dt * 16 + l16) * SEQ + s0 + (half ? 16 : 0);
            v16bf bv = load_frag2(vr, vr + 8);
            if (dt == 0)      O0 = WMMA_BF16(ap, bv, O0);
            else if (dt == 1) O1 = WMMA_BF16(ap, bv, O1);
            else if (dt == 2) O2 = WMMA_BF16(ap, bv, O2);
            else              O3 = WMMA_BF16(ap, bv, O3);
        }
    }

    // ---- epilogue: normalize and store to [B,T,H*D] bf16 ----
#pragma unroll
    for (int dt = 0; dt < 4; ++dt) {
        v8f Ov = (dt == 0) ? O0 : (dt == 1) ? O1 : (dt == 2) ? O2 : O3;
#pragma unroll
        for (int r = 0; r < 8; ++r) {
            int t = t0 + r + 8 * half;
            Ob[(b * SEQ + t) * N_EMBD + h * HEAD_DIM + dt * 16 + l16] =
                (__bf16)(Ov[r] / lrow[r]);
        }
    }
}

// ---------------------------------------------------------------------------
extern "C" void kernel_launch(void* const* d_in, const int* in_sizes, int n_in,
                              void* d_out, int out_size, void* d_ws, size_t ws_size,
                              hipStream_t stream) {
    const float* x  = (const float*)d_in[0];
    const float* Wq = (const float*)d_in[1];
    const float* Wk = (const float*)d_in[2];
    const float* Wv = (const float*)d_in[3];
    const float* Wo = (const float*)d_in[4];

    char* w = (char*)d_ws;
    __bf16* xb  = (__bf16*)w; w += (size_t)NROW * N_EMBD * 2;     // 16 MiB
    __bf16* Wqb = (__bf16*)w; w += (size_t)N_EMBD * N_EMBD * 2;   //  2 MiB
    __bf16* Wkb = (__bf16*)w; w += (size_t)N_EMBD * N_EMBD * 2;
    __bf16* Wvb = (__bf16*)w; w += (size_t)N_EMBD * N_EMBD * 2;
    __bf16* Wob = (__bf16*)w; w += (size_t)N_EMBD * N_EMBD * 2;
    __bf16* qb  = (__bf16*)w; w += (size_t)NROW * N_EMBD * 2;     // [B,H,T,D]
    __bf16* kb  = (__bf16*)w; w += (size_t)NROW * N_EMBD * 2;     // [B,H,T,D]
    __bf16* vtb = (__bf16*)w; w += (size_t)NROW * N_EMBD * 2;     // [B,H,D,T]
    __bf16* ob  = (__bf16*)w; w += (size_t)NROW * N_EMBD * 2;     // [B,T,C]

    // 1) fp32 -> bf16
    csa_f32_to_bf16<<<2048, 256, 0, stream>>>(x,  xb,  NROW * N_EMBD);
    csa_f32_to_bf16<<<1024, 256, 0, stream>>>(Wq, Wqb, N_EMBD * N_EMBD);
    csa_f32_to_bf16<<<1024, 256, 0, stream>>>(Wk, Wkb, N_EMBD * N_EMBD);
    csa_f32_to_bf16<<<1024, 256, 0, stream>>>(Wv, Wvb, N_EMBD * N_EMBD);
    csa_f32_to_bf16<<<1024, 256, 0, stream>>>(Wo, Wob, N_EMBD * N_EMBD);

    // 2) projections (q scaled by 1/sqrt(D); v stored transposed)
    dim3 gg(NROW / 64, N_EMBD / 64), bb(128);
    csa_gemm_xwt<<<gg, bb, 0, stream>>>(xb, Wqb, qb,  0, 0.125f);
    csa_gemm_xwt<<<gg, bb, 0, stream>>>(xb, Wkb, kb,  0, 1.0f);
    csa_gemm_xwt<<<gg, bb, 0, stream>>>(xb, Wvb, vtb, 1, 1.0f);

    // 3) causal flash attention
    csa_flash_attn<<<dim3(BATCH * N_HEAD, SEQ / 64), bb, 0, stream>>>(qb, kb, vtb, ob);

    // 4) output projection -> f32
    csa_gemm_xwt<<<gg, bb, 0, stream>>>(ob, Wob, d_out, 2, 1.0f);
}
